// Attention_62706522521647
// MI455X (gfx1250) — matrix operand order repaired and checked by reading, not compiled, not run on hardware
//
#include <hip/hip_runtime.h>
#include <math.h>

// ---------------------------------------------------------------------------
// Attention (B=4,N=2048,H=8,DQ=DV=48) for gfx1250.
// Compute-bound (~35 GFLOP vs ~25MB @ 23.3TB/s): all GEMMs on the f16 WMMA
// path (v_wmma_f32_16x16x32_f16, f32 accumulate). Flash-attention kernel uses
// 4-wave workgroups that stage shared K/V chunks into LDS via the CDNA5
// async-to-LDS path (ASYNCcnt) with double buffering, plus GL2 prefetch.
// ---------------------------------------------------------------------------

typedef __attribute__((ext_vector_type(8)))  float    v8f;
typedef __attribute__((ext_vector_type(16))) _Float16 v16h;
typedef __attribute__((ext_vector_type(8)))  _Float16 v8h;
typedef int v4i __attribute__((vector_size(16)));

union H16 { v16h v; v8h h[2]; };

#define WMMA_F32_F16(a, b, c) \
  __builtin_amdgcn_wmma_f32_16x16x32_f16(false, (a), false, (b), (short)0, (c), false, false)

__device__ inline v8f zero8() {
  v8f z = {0.f, 0.f, 0.f, 0.f, 0.f, 0.f, 0.f, 0.f};
  return z;
}

// ---- CDNA5 async global->LDS copy (ASYNCcnt) with portable fallback --------
#if defined(__gfx1250__) && __has_builtin(__builtin_amdgcn_global_load_async_to_lds_b128)
#define HAS_ASYNC_LDS 1
#else
#define HAS_ASYNC_LDS 0
#endif

__device__ inline void async_copy16(const _Float16* g, _Float16* l) {
#if HAS_ASYNC_LDS
  __builtin_amdgcn_global_load_async_to_lds_b128(
      (__attribute__((address_space(1))) v4i*)g,
      (__attribute__((address_space(3))) v4i*)l, 0, 0);
#else
  *(v8h*)l = *(const v8h*)g;   // through VGPRs
#endif
}

__device__ inline void async_wait() {
#if HAS_ASYNC_LDS
#if __has_builtin(__builtin_amdgcn_s_wait_asynccnt)
  __builtin_amdgcn_s_wait_asynccnt(0);
#else
  asm volatile("s_wait_asynccnt 0" ::: "memory");
#endif
#endif
}

constexpr int B_   = 4;
constexpr int N_   = 2048;
constexpr int H_   = 8;
constexpr int DQ_  = 48;
constexpr int DV_  = 48;
constexpr int IND  = H_ * DQ_;          // 384
constexpr int KVW  = H_ * (DQ_ + DV_);  // 768
constexpr int DPAD = 64;                // q/k feature dim padded 48 -> 64 (zeros)

// ------------------------------ prep: f32 -> f16 casts + weight transposes ---
__global__ void prep_kernel(const float* __restrict__ x,
                            const float* __restrict__ Wq,
                            const float* __restrict__ Wkv,
                            const float* __restrict__ Wout,
                            _Float16* __restrict__ xh,
                            _Float16* __restrict__ WqT,
                            _Float16* __restrict__ WkvT,
                            _Float16* __restrict__ WoutT) {
  const long tid    = (long)blockIdx.x * blockDim.x + threadIdx.x;
  const long stride = (long)gridDim.x * blockDim.x;
  const long XE = (long)B_ * N_ * IND;
  for (long t = tid; t < XE; t += stride) xh[t] = (_Float16)x[t];
  for (long t = tid; t < (long)IND * IND; t += stride) {
    long k = t / IND, c = t % IND;
    WqT[c * IND + k] = (_Float16)Wq[t];
  }
  for (long t = tid; t < (long)IND * KVW; t += stride) {
    long k = t / KVW, c = t % KVW;
    WkvT[c * IND + k] = (_Float16)Wkv[t];
  }
  for (long t = tid; t < (long)IND * IND; t += stride) {
    long k = t / IND, c = t % IND;
    WoutT[c * IND + k] = (_Float16)Wout[t];
  }
}

// -------- QKV projection (WMMA) + fused L2-norm + RoPE + layout conversion ---
__global__ void __launch_bounds__(32)
qkv_kernel(const _Float16* __restrict__ xh,
           const _Float16* __restrict__ WqT,
           const _Float16* __restrict__ WkvT,
           const float* __restrict__ pos,
           const float* __restrict__ scale,
           _Float16* __restrict__ qh,   // [B][H][N][64]
           _Float16* __restrict__ kh,   // [B][H][N][64]
           _Float16* __restrict__ vT) { // [B][H][48][N]
  __shared__ float ldsS[16][144];       // 16 tokens x (q48 | k48 | v48), f32

  const int lane = threadIdx.x;
  const int n    = lane & 15;
  const int grp  = lane >> 4;
  const int tile = blockIdx.x;          // 0..511
  const int h    = blockIdx.y;          // 0..7
  const int b    = tile >> 7;           // 128 tiles per batch
  const int t0   = (tile & 127) * 16;

  const long rowA = (long)(b * N_ + t0 + n) * IND;  // A: row m = lane%16

  v8f acc[9];
#pragma unroll
  for (int i = 0; i < 9; ++i) acc[i] = zero8();

  for (int kc = 0; kc < 12; ++kc) {
    H16 a;
    const _Float16* ap = xh + rowA + kc * 32 + grp * 16;
    a.h[0] = *(const v8h*)ap;          // K = grp*16 .. +7
    a.h[1] = *(const v8h*)(ap + 8);    // K = grp*16+8 .. +7
#pragma unroll
    for (int nt = 0; nt < 9; ++nt) {
      const _Float16* WT;
      long col;
      if (nt < 3)      { WT = WqT;  col = (long)h * 48 + nt * 16 + n; }
      else if (nt < 6) { WT = WkvT; col = (long)h * 96 + (nt - 3) * 16 + n; }
      else             { WT = WkvT; col = (long)h * 96 + 48 + (nt - 6) * 16 + n; }
      v16h bv = *(const v16h*)(WT + col * IND + kc * 32 + grp * 16);
      acc[nt] = WMMA_F32_F16(a.v, bv, acc[nt]);
    }
  }

  // Spill accumulators (C/D layout: lane=n, vgpr r -> m = r + 8*grp) to LDS.
#pragma unroll
  for (int nt = 0; nt < 9; ++nt)
#pragma unroll
    for (int r = 0; r < 8; ++r)
      ldsS[r + 8 * grp][nt * 16 + n] = acc[nt][r];
  __syncthreads();

  // lanes 0-15: q row of token n; lanes 16-31: k row of token n.
  const int   tok  = t0 + n;
  const float sc   = sqrtf(scale[h]);
  const float* rowp = &ldsS[n][grp ? 48 : 0];

  float ss = 0.f;
#pragma unroll
  for (int i = 0; i < 48; ++i) ss += rowp[i] * rowp[i];
  const float mult = sc * rsqrtf(ss + 1e-6f);

  const float* pp = pos + ((long)b * N_ + tok) * 24;
  _Float16* dst = (grp ? kh : qh) + ((long)(b * H_ + h) * N_ + tok) * DPAD;

  const float LOGPI = 1.14472988584940017f;         // ln(pi)
  const float STEP  = 2.30258509299404568f / 192.f; // ln(10)/192
#pragma unroll
  for (int d = 0; d < 24; ++d) {
    float th = pp[d] * __expf(LOGPI + (float)(d * 8 + h) * STEP);
    float s, c;
    __sincosf(th, &s, &c);
    float x1 = rowp[d] * mult, x2 = rowp[24 + d] * mult;
    dst[d]      = (_Float16)(x1 * c - x2 * s);
    dst[24 + d] = (_Float16)(x2 * c + x1 * s);
  }
#pragma unroll
  for (int i = 48; i < 64; ++i) dst[i] = (_Float16)0.f;

  // v: copy transposed so PV B-operand is contiguous along keys.
  for (int dv = lane; dv < 48; dv += 32) {
    _Float16* vd = vT + ((long)(b * H_ + h) * 48 + dv) * N_ + t0;
#pragma unroll
    for (int t = 0; t < 16; ++t) vd[t] = (_Float16)ldsS[t][96 + dv];
  }
}

// ----------------- flash attention ------------------------------------------
// 4 waves per block, each owning a 16-query tile of the same (b,h).
// K/V chunks (32 keys) are staged cooperatively into double-buffered LDS via
// async-to-LDS; waves compute WMMAs out of LDS while the next chunk streams in.
__global__ void __launch_bounds__(128)
attn_kernel(const _Float16* __restrict__ qh,
            const _Float16* __restrict__ kh,
            const _Float16* __restrict__ vT,
            _Float16* __restrict__ oh) {  // [B][N][H*48]
  __shared__ __align__(16) _Float16 ldsK[2][32][64];  // [buf][key][feat]
  __shared__ __align__(16) _Float16 ldsV[2][48][32];  // [buf][dv][key]
  __shared__ __align__(32) _Float16 ldsP[4][16][32];  // per-wave P transpose

  const int tid  = threadIdx.x;
  const int wid  = tid >> 5;
  const int lane = tid & 31;
  const int n    = lane & 15;
  const int grp  = lane >> 4;
  const int bh   = blockIdx.y;   // 0..31
  const int b    = bh / H_;
  const int h    = bh % H_;
  const int t0   = (blockIdx.x * 4 + wid) * 16;

  const _Float16* kbase0 = kh + (long)bh * N_ * DPAD;
  const _Float16* vbase0 = vT + (long)bh * 48 * N_;

  // Q tile A-operands (features 0-31 and 32-63), loaded once per wave.
  const _Float16* qrow = qh + ((long)bh * N_ + t0 + n) * DPAD;
  H16 qa0, qa1;
  qa0.h[0] = *(const v8h*)(qrow + grp * 16);
  qa0.h[1] = *(const v8h*)(qrow + 8 + grp * 16);
  qa1.h[0] = *(const v8h*)(qrow + 32 + grp * 16);
  qa1.h[1] = *(const v8h*)(qrow + 40 + grp * 16);

  // Cooperative stage of one 32-key chunk (K: 4KB, V: 3KB) into buffer bb.
  auto stage = [&](int kc, int bb) {
    for (int u = tid; u < 256; u += 128) {        // K: 32 rows x 8 units
      int row = u >> 3, seg = u & 7;
      async_copy16(kbase0 + (long)(kc + row) * DPAD + seg * 8,
                   &ldsK[bb][row][seg * 8]);
    }
    for (int u = tid; u < 192; u += 128) {        // V: 48 rows x 4 units
      int row = u >> 2, seg = u & 3;
      async_copy16(vbase0 + (long)row * N_ + kc + seg * 8,
                   &ldsV[bb][row][seg * 8]);
    }
  };

  float mrun[8], lrun[8];
  v8f o0 = zero8(), o1 = zero8(), o2 = zero8();
#pragma unroll
  for (int r = 0; r < 8; ++r) { mrun[r] = -1e30f; lrun[r] = 0.f; }

  stage(0, 0);
  async_wait();
  __syncthreads();

  const int NCH = N_ / 32;  // 64 chunks
  for (int ci = 0; ci < NCH; ++ci) {
    const int bb = ci & 1;
    if (ci + 1 < NCH) stage((ci + 1) * 32, bb ^ 1);
    if (ci + 2 < NCH)  // keep the K stream warm in GL2 (global_prefetch_b8)
      __builtin_prefetch(kbase0 + (long)((ci + 2) * 32 + lane) * DPAD, 0, 1);

    // S = Q K^T for keys [ci*32, ci*32+32), K = 64 padded features.
    v8f S0 = zero8(), S1 = zero8();
    v16h kb;
    kb = *(const v16h*)&ldsK[bb][n][grp * 16];
    S0 = WMMA_F32_F16(qa0.v, kb, S0);
    kb = *(const v16h*)&ldsK[bb][n][32 + grp * 16];
    S0 = WMMA_F32_F16(qa1.v, kb, S0);
    kb = *(const v16h*)&ldsK[bb][16 + n][grp * 16];
    S1 = WMMA_F32_F16(qa0.v, kb, S1);
    kb = *(const v16h*)&ldsK[bb][16 + n][32 + grp * 16];
    S1 = WMMA_F32_F16(qa1.v, kb, S1);

    // Online softmax. C/D rows live on 16-lane groups; xor masks 1/2/4/8
    // reduce across the N dimension without leaving the group.
#pragma unroll
    for (int r = 0; r < 8; ++r) {
      float s0 = S0[r], s1 = S1[r];
      float mx = fmaxf(s0, s1);
      mx = fmaxf(mx, __shfl_xor(mx, 1));
      mx = fmaxf(mx, __shfl_xor(mx, 2));
      mx = fmaxf(mx, __shfl_xor(mx, 4));
      mx = fmaxf(mx, __shfl_xor(mx, 8));
      float mnew  = fmaxf(mrun[r], mx);
      float alpha = __expf(mrun[r] - mnew);
      float p0 = __expf(s0 - mnew);
      float p1 = __expf(s1 - mnew);
      float rs = p0 + p1;
      rs += __shfl_xor(rs, 1);
      rs += __shfl_xor(rs, 2);
      rs += __shfl_xor(rs, 4);
      rs += __shfl_xor(rs, 8);
      mrun[r] = mnew;
      lrun[r] = lrun[r] * alpha + rs;
      o0[r] *= alpha;
      o1[r] *= alpha;
      o2[r] *= alpha;
      ldsP[wid][r + 8 * grp][n]      = (_Float16)p0;  // D-layout -> LDS
      ldsP[wid][r + 8 * grp][16 + n] = (_Float16)p1;
    }
    // Per-wave LDS round-trip: DS ops are in-order within a wave, no barrier.
    H16 pa;
    pa.h[0] = *(const v8h*)&ldsP[wid][n][grp * 16];
    pa.h[1] = *(const v8h*)&ldsP[wid][n][8 + grp * 16];

    v16h vb;
    vb = *(const v16h*)&ldsV[bb][n][grp * 16];
    o0 = WMMA_F32_F16(pa.v, vb, o0);
    vb = *(const v16h*)&ldsV[bb][16 + n][grp * 16];
    o1 = WMMA_F32_F16(pa.v, vb, o1);
    vb = *(const v16h*)&ldsV[bb][32 + n][grp * 16];
    o2 = WMMA_F32_F16(pa.v, vb, o2);

    async_wait();     // our share of the next chunk has landed in LDS
    __syncthreads();  // all waves done with buf bb + staging of bb^1
  }

#pragma unroll
  for (int r = 0; r < 8; ++r) {
    float inv = 1.0f / lrun[r];
    int m = r + 8 * grp;
    _Float16* orow = oh + ((long)(b * N_ + t0 + m)) * IND + h * 48;
    orow[n]      = (_Float16)(o0[r] * inv);
    orow[16 + n] = (_Float16)(o1[r] * inv);
    orow[32 + n] = (_Float16)(o2[r] * inv);
  }
}

// -------------------------- output projection (WMMA), f32 result ------------
__global__ void __launch_bounds__(32)
outproj_kernel(const _Float16* __restrict__ oh,
               const _Float16* __restrict__ WoutT,
               float* __restrict__ out) {
  const int lane = threadIdx.x;
  const int n    = lane & 15;
  const int grp  = lane >> 4;
  const int rt   = blockIdx.x;   // 0..511 (16-row tiles over B*N)
  const int cg   = blockIdx.y;   // 0..3   (96 output cols each)

  const long row = (long)rt * 16 + n;
  v8f acc[6];
#pragma unroll
  for (int j = 0; j < 6; ++j) acc[j] = zero8();

  for (int kc = 0; kc < 12; ++kc) {
    H16 a;
    const _Float16* ap = oh + row * IND + kc * 32 + grp * 16;
    a.h[0] = *(const v8h*)ap;
    a.h[1] = *(const v8h*)(ap + 8);
#pragma unroll
    for (int j = 0; j < 6; ++j) {
      long col = (long)cg * 96 + j * 16 + n;
      v16h bv = *(const v16h*)(WoutT + col * IND + kc * 32 + grp * 16);
      acc[j] = WMMA_F32_F16(a.v, bv, acc[j]);
    }
  }
#pragma unroll
  for (int j = 0; j < 6; ++j)
#pragma unroll
    for (int r = 0; r < 8; ++r) {
      int m = r + 8 * grp;
      out[((long)rt * 16 + m) * IND + cg * 96 + j * 16 + n] = acc[j][r];
    }
}

// ---------------------------------------------------------------------------
extern "C" void kernel_launch(void* const* d_in, const int* in_sizes, int n_in,
                              void* d_out, int out_size, void* d_ws, size_t ws_size,
                              hipStream_t stream) {
  const float* x     = (const float*)d_in[0];
  const float* pos   = (const float*)d_in[1];
  const float* Wq    = (const float*)d_in[2];
  const float* Wkv   = (const float*)d_in[3];
  const float* Wout  = (const float*)d_in[4];
  const float* scale = (const float*)d_in[5];
  float* out = (float*)d_out;

  char* ws   = (char*)d_ws;
  size_t off = 0;
  auto take = [&](size_t bytes) -> _Float16* {
    char* p = ws + off;
    off += (bytes + 255) & ~(size_t)255;
    return (_Float16*)p;
  };
  _Float16* xh    = take((size_t)B_ * N_ * IND * 2);        // 6.3 MB
  _Float16* WqT   = take((size_t)IND * IND * 2);
  _Float16* WkvT  = take((size_t)KVW * IND * 2);
  _Float16* WoutT = take((size_t)IND * IND * 2);
  _Float16* qh    = take((size_t)B_ * H_ * N_ * DPAD * 2);  // 8.4 MB
  _Float16* kh    = take((size_t)B_ * H_ * N_ * DPAD * 2);  // 8.4 MB
  _Float16* vT    = take((size_t)B_ * H_ * 48 * N_ * 2);    // 6.3 MB
  _Float16* oh    = take((size_t)B_ * N_ * IND * 2);        // 6.3 MB
  (void)ws_size; (void)in_sizes; (void)n_in; (void)out_size;

  prep_kernel<<<1024, 256, 0, stream>>>(x, Wq, Wkv, Wout, xh, WqT, WkvT, WoutT);
  qkv_kernel<<<dim3(512, 8), 32, 0, stream>>>(xh, WqT, WkvT, pos, scale, qh, kh, vT);
  attn_kernel<<<dim3(32, 32), 128, 0, stream>>>(qh, kh, vT, oh);
  outproj_kernel<<<dim3(512, 4), 32, 0, stream>>>(oh, WoutT, out);
}
